// CTGCalibratedBinary_91113436217701
// MI455X (gfx1250) — matrix operand order repaired and checked by reading, not compiled, not run on hardware
//
#include <hip/hip_runtime.h>
#include <hip/hip_bf16.h>

// ---------------------------------------------------------------------------
// out[8192,4096] = x[8192,4096] @ (sign * block_mag), block_mag constant per
// 64x64 block.  Factored as per-64K-chunk bf16 WMMA + fp32 scale-accumulate.
// ---------------------------------------------------------------------------

typedef __attribute__((ext_vector_type(16))) __bf16 v16bf;
typedef __attribute__((ext_vector_type(8)))  __bf16 v8bf;
typedef __attribute__((ext_vector_type(8)))  float  v8f;

#define N_IN   4096
#define N_OUT  4096
#define TOKENS 8192
#define NBO    (N_OUT / 64)   // block-scale columns

#define MT  128   // C-tile rows per block
#define NT  128   // C-tile cols per block
#define KT  64    // K chunk (== scale block size)
#define LDA 72    // padded LDS stride (bf16 elems) for A tile
#define LDB 72    // padded LDS stride (bf16 elems) for B tile

__device__ __forceinline__ v16bf cat16(v8bf lo, v8bf hi) {
  return __builtin_shufflevector(lo, hi, 0, 1, 2, 3, 4, 5, 6, 7,
                                          8, 9, 10, 11, 12, 13, 14, 15);
}

__global__ __launch_bounds__(256)
void ctg_binary_gemm_kernel(const float* __restrict__ x,
                            const float* __restrict__ sign,
                            const float* __restrict__ scales,
                            float* __restrict__ out)
{
  __shared__ __bf16 As[MT * LDA];   // x tile, row-major, K contiguous
  __shared__ __bf16 Bs[NT * LDB];   // sign tile, transposed (n-major, K contiguous)

  const int tid  = threadIdx.x;
  const int lane = tid & 31;
  const int wave = tid >> 5;
  const int wm   = wave >> 2;       // 0..1 : 64-row slice
  const int wn   = wave & 3;        // 0..3 : 32-col slice
  const int g    = lane >> 4;       // lane group (0/1)
  const int lm   = lane & 15;       // lane within group

  const int block_m = blockIdx.y * MT;
  const int block_n = blockIdx.x * NT;

  // running fp32 accumulators: 4 M-subtiles x 2 N-subtiles of 16x16
  v8f acc[4][2];
#pragma unroll
  for (int i = 0; i < 4; ++i)
#pragma unroll
    for (int j = 0; j < 2; ++j)
      acc[i][j] = (v8f)0.0f;

  // both 16-wide N subtiles of this wave sit inside one 64-wide scale block
  const int nb = (block_n + wn * 32) >> 6;

  for (int kb = 0; kb < N_IN / KT; ++kb) {
    const int k0 = kb * KT;
    __syncthreads();

    // ---- stage A tile: 128 rows x 64 K of x, fp32 -> bf16 ----------------
    {
      const int rg = tid >> 4;            // 16 rows per pass
      const int cg = tid & 15;            // float4 index within the 64-K row
#pragma unroll
      for (int r = rg; r < MT; r += 16) {
        const float4 v = *(const float4*)(x + (size_t)(block_m + r) * N_IN + k0 + cg * 4);
        union { __bf16 h[4]; uint2 u; } p;
        p.h[0] = (__bf16)v.x; p.h[1] = (__bf16)v.y;
        p.h[2] = (__bf16)v.z; p.h[3] = (__bf16)v.w;
        *(uint2*)(&As[r * LDA + cg * 4]) = p.u;
      }
    }

    // ---- stage B tile: 64 K-rows x 128 N of sign, transposed into LDS ----
    {
      const int kr = tid >> 5;            // 8 K-rows per pass
      const int nq = tid & 31;            // float4 index along N (32*4 = 128)
#pragma unroll
      for (int k = kr; k < KT; k += 8) {
        const float4 v = *(const float4*)(sign + (size_t)(k0 + k) * N_OUT + block_n + nq * 4);
        Bs[(nq * 4 + 0) * LDB + k] = (__bf16)v.x;
        Bs[(nq * 4 + 1) * LDB + k] = (__bf16)v.y;
        Bs[(nq * 4 + 2) * LDB + k] = (__bf16)v.z;
        Bs[(nq * 4 + 3) * LDB + k] = (__bf16)v.w;
      }
    }

    __syncthreads();

    // per-(K-chunk, N-block) magnitude, kept in fp32
    float sc = scales[kb * NBO + nb];
    sc = fmaxf(sc, 0.0f) + 1e-6f;

    // ---- WMMA compute over this K chunk ----------------------------------
    v8f part[4][2];
#pragma unroll
    for (int i = 0; i < 4; ++i)
#pragma unroll
      for (int j = 0; j < 2; ++j)
        part[i][j] = (v8f)0.0f;

#pragma unroll
    for (int s = 0; s < 2; ++s) {        // two k=32 WMMA steps
      const int ko = s * 32;

      v16bf a[4];
#pragma unroll
      for (int i = 0; i < 4; ++i) {
        const __bf16* base = &As[(wm * 64 + i * 16 + lm) * LDA + ko];
        // A 16x32 bf16 layout: elems 0..7 = K[g*8 .. g*8+7], 8..15 = K[16+g*8 ..]
        a[i] = cat16(*(const v8bf*)(base + g * 8),
                     *(const v8bf*)(base + 16 + g * 8));
      }

      v16bf b[2];
#pragma unroll
      for (int j = 0; j < 2; ++j) {
        const __bf16* base = &Bs[(wn * 32 + j * 16 + lm) * LDB + ko];
        // B 32x16 bf16 layout: elems 0..7 = K[g*8 .. g*8+7], 8..15 = K[16+g*8 ..] for column lm
        b[j] = cat16(*(const v8bf*)(base + g * 8),
                     *(const v8bf*)(base + 16 + g * 8));
      }

#pragma unroll
      for (int i = 0; i < 4; ++i)
#pragma unroll
        for (int j = 0; j < 2; ++j)
          part[i][j] = __builtin_amdgcn_wmma_f32_16x16x32_bf16(
              /*neg_a=*/false, a[i], /*neg_b=*/false, b[j],
              /*c_mod=*/(short)0, part[i][j],
              /*reuse_a=*/false, /*reuse_b=*/false);
    }

    // ---- fp32 scale-accumulate (exact block-magnitude handling) ----------
#pragma unroll
    for (int i = 0; i < 4; ++i)
#pragma unroll
      for (int j = 0; j < 2; ++j)
#pragma unroll
        for (int e = 0; e < 8; ++e)
          acc[i][j][e] += sc * part[i][j][e];
  }

  // ---- writeout: C/D layout = lane(g,lm), VGPR r -> (M = r + 8g, N = lm) --
#pragma unroll
  for (int i = 0; i < 4; ++i) {
#pragma unroll
    for (int j = 0; j < 2; ++j) {
      const int row0 = block_m + wm * 64 + i * 16 + 8 * g;
      const int col  = block_n + wn * 32 + j * 16 + lm;
      float* o = out + (size_t)row0 * N_OUT + col;
#pragma unroll
      for (int r = 0; r < 8; ++r)
        o[(size_t)r * N_OUT] = acc[i][j][r];
    }
  }
}

extern "C" void kernel_launch(void* const* d_in, const int* in_sizes, int n_in,
                              void* d_out, int out_size, void* d_ws, size_t ws_size,
                              hipStream_t stream) {
  const float* x      = (const float*)d_in[0];   // (8192, 4096) fp32
  const float* sign   = (const float*)d_in[1];   // (4096, 4096) fp32 (+/-1)
  const float* scales = (const float*)d_in[2];   // (64, 64) fp32
  float* out          = (float*)d_out;           // (8192, 4096) fp32

  dim3 grid(N_OUT / NT, TOKENS / MT);            // (32, 64)
  dim3 block(256);
  ctg_binary_gemm_kernel<<<grid, block, 0, stream>>>(x, sign, scales, out);
}
